// Tacotron_47957604827592
// MI455X (gfx1250) — compile-verified, matrix operand order checked by reading, not run on hardware
//
#include <hip/hip_runtime.h>
#include <math.h>
#include <stdint.h>

typedef __attribute__((ext_vector_type(16))) __bf16 v16bf;
typedef __attribute__((ext_vector_type(8)))  float  v8f;

__device__ __forceinline__ int koffs(int j, int half) {
    int vg = j >> 1, lo = j & 1;
    return ((vg & 4) ? 16 : 0) + (half ? 8 : 0) + ((vg & 3) << 1) + lo;
}
// inverse of koffs: for k-in-chunk kk, which (half, element j) holds it
__device__ __forceinline__ void fragPos(int kk, int& half, int& j) {
    half = (kk >> 3) & 1;
    int vg = ((kk & 16) >> 2) + ((kk & 7) >> 1);
    j = (vg << 1) + (kk & 1);
}
__device__ __forceinline__ float sigmf(float x) { return 1.0f / (1.0f + __expf(-x)); }

// ---------------------------------------------------------------------------
// Pack a weight matrix into WMMA B-fragment order (bf16).
// dst tile (nt, kt): 32 lanes x 16 bf16; element j of lane L = W^T[k][n],
// n = nt*16 + L%16, k = (kt-ktOff)*32 + koffs(j, L/16) (k relative to src).
// mode 0: src [N][K] row-major (x @ W.T);  mode 1: src [K][N].
// ---------------------------------------------------------------------------
__global__ void k_pack(const float* __restrict__ src, int mode, int Ksrc, int Nsrc,
                       __bf16* __restrict__ dst, int nKtTot, int ktOff, int ktCnt,
                       int Ntiles)
{
    int idx = blockIdx.x * 256 + threadIdx.x;
    int total = Ntiles * ktCnt * 32;
    if (idx >= total) return;
    int lane = idx & 31;
    int tile = idx >> 5;
    int ktl  = tile % ktCnt;
    int nt   = tile / ktCnt;
    int n    = nt * 16 + (lane & 15);
    int half = lane >> 4;
    size_t base = ((size_t)(nt * nKtTot + (ktOff + ktl)) * 32 + lane) * 16;
#pragma unroll
    for (int j = 0; j < 16; ++j) {
        int k = ktl * 32 + koffs(j, half);
        float v = 0.0f;
        if (k < Ksrc && n < Nsrc)
            v = (mode == 0) ? src[(size_t)n * Ksrc + k] : src[(size_t)k * Nsrc + n];
        dst[base + j] = (__bf16)v;
    }
}

// ---------------------------------------------------------------------------
// Generic setup GEMM: out[R x N] = act(A[R x K] * Wpacked + bias)
// grid (R/64, N/32), block 256 (8 waves). Optional outputs: f32 row-major,
// bf16 row-major, or bf16 A-fragment stream for a downstream consumer whose
// 64-row block == blockIdx.x (used to pre-fragment the prenet outputs).
// ---------------------------------------------------------------------------
template <typename TA>
__global__ void k_gemm(const TA* __restrict__ A, int K,
                       const __bf16* __restrict__ Wp, int N,
                       const float* __restrict__ bias, int relu,
                       float* __restrict__ outF, __bf16* __restrict__ outH,
                       __bf16* __restrict__ outFrag, int fragNKt)
{
    __shared__ __align__(32) __bf16 sX[2048];
    int tid = threadIdx.x;
    int wave = tid >> 5, lane = tid & 31;
    int m_t = wave >> 1, n_sub = wave & 1;
    int rowBase = blockIdx.x * 64;
    int nKt = K >> 5;
    int nt = blockIdx.y * 2 + n_sub;
    v8f acc = {};
    for (int kt = 0; kt < nKt; ++kt) {
        int k0 = kt << 5;
#pragma unroll
        for (int e = 0; e < 8; ++e) {
            int fid = e * 256 + tid;
            int fl  = (fid >> 4) & 31;
            int row = ((fid >> 9) << 4) + (fl & 15);
            int kk  = k0 + koffs(fid & 15, fl >> 4);
            sX[fid] = (__bf16)(float)A[(size_t)(rowBase + row) * K + kk];
        }
        __syncthreads();
        v16bf a = *(const v16bf*)&sX[(m_t * 32 + lane) * 16];
        v16bf b = *(const v16bf*)(Wp + ((size_t)(nt * nKt + kt) * 32 + lane) * 16);
        acc = __builtin_amdgcn_wmma_f32_16x16x32_bf16(false, a, false, b, (short)0,
                                                      acc, false, false);
        __syncthreads();
    }
    int n = blockIdx.y * 32 + n_sub * 16 + (lane & 15);
    int halfD = lane >> 4;
    float bv = bias ? bias[n] : 0.0f;
    int hf, jf;
    fragPos(n & 31, hf, jf);
    int ktA = n >> 5;
#pragma unroll
    for (int r = 0; r < 8; ++r) {
        int m64 = m_t * 16 + halfD * 8 + r;
        float v = acc[r] + bv;
        if (relu) v = fmaxf(v, 0.0f);
        size_t oi = (size_t)(rowBase + m64) * N + n;
        if (outF) outF[oi] = v;
        if (outH) outH[oi] = (__bf16)v;
        if (outFrag) {
            int laneF = hf * 16 + (m64 & 15);
            size_t fi = ((size_t)blockIdx.x * fragNKt + ktA) * 2048 +
                        (size_t)(m64 >> 4) * 512 + laneF * 16 + jf;
            outFrag[fi] = (__bf16)v;
        }
    }
}

// ---------------------------------------------------------------------------
// Fused LSTM gates + cell update; all operands are pre-fragmented bf16
// streams (2048 bf16 = one 64x32 A-fragment chunk). Double-buffered
// async-to-LDS pipeline: copy of chunk k+1 overlaps WMMAs of chunk k.
// grid 64 blocks x 128 threads (4 waves = 4 M-tiles; 16 cols/gate/block).
// ---------------------------------------------------------------------------
__global__ void k_lstm(const __bf16* __restrict__ fragA, int ncA,
                       const __bf16* __restrict__ fragB, int ncB,
                       const __bf16* __restrict__ fragH,
                       const __bf16* __restrict__ Wp, int nKt,
                       const float* __restrict__ bias,
                       float* __restrict__ c, float* __restrict__ hOut,
                       __bf16* __restrict__ hFragOut)
{
    __shared__ __align__(32) __bf16 sX[2][2048];
    int tid = threadIdx.x;
    int wave = tid >> 5, lane = tid & 31;
    int m_t = wave;                 // 4 waves = 4 M tiles
    int ntb = blockIdx.x;           // 16 columns of each gate region
    unsigned ldsBase = (unsigned)(uintptr_t)(void*)&sX[0][0];
    v8f acc0 = {}, acc1 = {}, acc2 = {}, acc3 = {};

    auto srcFor = [&](int kt) -> const __bf16* {
        const __bf16* s = fragA + (size_t)kt * 2048;
        if (kt >= ncA)       s = fragB + (size_t)(kt - ncA) * 2048;
        if (kt >= ncA + ncB) s = fragH + (size_t)(kt - ncA - ncB) * 2048;
        return s;
    };
    auto issue = [&](int kt, int buf) {
        const __bf16* g = srcFor(kt) + tid * 16;   // 32 bytes per thread
        unsigned lds = ldsBase + (unsigned)buf * 4096u + (unsigned)tid * 32u;
        asm volatile("global_load_async_to_lds_b128 %0, %1, off"
                     :: "v"(lds), "v"(g) : "memory");
        asm volatile("global_load_async_to_lds_b128 %0, %1, off"
                     :: "v"(lds + 16u), "v"(g + 8) : "memory");
    };

    issue(0, 0);
    for (int kt = 0; kt < nKt; ++kt) {
        if (kt + 1 < nKt) {
            issue(kt + 1, (kt + 1) & 1);
            // 2 copies just issued for k+1; wait until only those remain.
            asm volatile("s_wait_asynccnt 0x2" ::: "memory");
        } else {
            asm volatile("s_wait_asynccnt 0x0" ::: "memory");
        }
        if (kt + 1 < nKt)
            __builtin_prefetch((const void*)(Wp + ((size_t)(ntb * nKt + kt + 1) * 32) * 16), 0, 1);
        __syncthreads();
        v16bf a = *(const v16bf*)&sX[kt & 1][(m_t * 32 + lane) * 16];
        const v16bf* b0 = (const v16bf*)(Wp + ((size_t)((  0 + ntb) * nKt + kt) * 32 + lane) * 16);
        const v16bf* b1 = (const v16bf*)(Wp + ((size_t)(( 64 + ntb) * nKt + kt) * 32 + lane) * 16);
        const v16bf* b2 = (const v16bf*)(Wp + ((size_t)((128 + ntb) * nKt + kt) * 32 + lane) * 16);
        const v16bf* b3 = (const v16bf*)(Wp + ((size_t)((192 + ntb) * nKt + kt) * 32 + lane) * 16);
        acc0 = __builtin_amdgcn_wmma_f32_16x16x32_bf16(false, a, false, *b0, (short)0, acc0, false, false);
        acc1 = __builtin_amdgcn_wmma_f32_16x16x32_bf16(false, a, false, *b1, (short)0, acc1, false, false);
        acc2 = __builtin_amdgcn_wmma_f32_16x16x32_bf16(false, a, false, *b2, (short)0, acc2, false, false);
        acc3 = __builtin_amdgcn_wmma_f32_16x16x32_bf16(false, a, false, *b3, (short)0, acc3, false, false);
        __syncthreads();
    }
    int n = ntb * 16 + (lane & 15);
    int halfD = lane >> 4;
    float bi  = bias[n];
    float bfv = bias[1024 + n];
    float bg  = bias[2048 + n];
    float bo  = bias[3072 + n];
    int hf, jf;
    fragPos(n & 31, hf, jf);
    int ktH = n >> 5;
#pragma unroll
    for (int r = 0; r < 8; ++r) {
        int m = m_t * 16 + halfD * 8 + r;
        size_t idx = (size_t)m * 1024 + n;
        float iv = sigmf(acc0[r] + bi);
        float fv = sigmf(acc1[r] + bfv);
        float gv = tanhf(acc2[r] + bg);
        float ov = sigmf(acc3[r] + bo);
        float cn = fv * c[idx] + iv * gv;
        c[idx] = cn;
        float hv = ov * tanhf(cn);
        hOut[idx] = hv;
        int laneF = hf * 16 + (m & 15);
        hFragOut[((size_t)ktH * 2048) + (size_t)(m >> 4) * 512 + laneF * 16 + jf] = (__bf16)hv;
    }
}

// ---------------------------------------------------------------------------
// Location-sensitive attention: one workgroup per batch row (owns the row,
// so in-place prev_w / cum_w updates are race-free). block 256 == L.
// Emits ctx both row-major f32 and as bf16 A-fragments for the LSTMs.
// ---------------------------------------------------------------------------
__global__ void k_attn(const float* __restrict__ hAtt, const float* __restrict__ enc,
                       const float* __restrict__ pmem, const int* __restrict__ lengths,
                       const float* __restrict__ qW, const float* __restrict__ locK,
                       const float* __restrict__ locW, const float* __restrict__ vvec,
                       float* __restrict__ prevW, float* __restrict__ cumW,
                       float* __restrict__ ctx, __bf16* __restrict__ ctxFrag,
                       float* __restrict__ wsOut, int t)
{
    __shared__ float sH[1024];
    __shared__ float sQ[128];
    __shared__ float sLW[4096];
    __shared__ float sLK[1984];
    __shared__ float sV[128];
    __shared__ float sPrev[256];
    __shared__ float sCum[256];
    __shared__ float sWt[256];
    __shared__ float sRed[256];
    int b = blockIdx.x, tid = threadIdx.x;
    for (int i = tid; i < 1024; i += 256) sH[i]  = hAtt[b * 1024 + i];
    for (int i = tid; i < 4096; i += 256) sLW[i] = locW[i];
    for (int i = tid; i < 1984; i += 256) sLK[i] = locK[i];
    if (tid < 128) sV[tid] = vvec[tid];
    sPrev[tid] = prevW[b * 256 + tid];
    sCum[tid]  = cumW[b * 256 + tid];
    __syncthreads();
    if (tid < 128) {
        float a = 0.f;
        for (int k = 0; k < 1024; ++k) a += sH[k] * qW[k * 128 + tid];
        sQ[tid] = a;
    }
    __syncthreads();
    int l = tid;
    float lf[32];
#pragma unroll
    for (int f = 0; f < 32; ++f) lf[f] = 0.f;
    for (int k = 0; k < 31; ++k) {
        int pos = l + k - 15;
        bool ok = (pos >= 0 && pos < 256);
        int pc = ok ? pos : 0;
        float pv = ok ? sPrev[pc] : 0.f;
        float cv = ok ? sCum[pc]  : 0.f;
#pragma unroll
        for (int f = 0; f < 32; ++f)
            lf[f] += sLK[(f * 2 + 0) * 31 + k] * pv + sLK[(f * 2 + 1) * 31 + k] * cv;
    }
    float e = 0.f;
    size_t pb = ((size_t)b * 256 + l) * 128;
    for (int a = 0; a < 128; ++a) {
        float s = sQ[a] + pmem[pb + a];
#pragma unroll
        for (int f = 0; f < 32; ++f) s += lf[f] * sLW[f * 128 + a];
        e += sV[a] * tanhf(s);
    }
    if (l >= lengths[b]) e = -1e9f;
    sRed[tid] = e; __syncthreads();
    for (int s = 128; s > 0; s >>= 1) {
        if (tid < s) sRed[tid] = fmaxf(sRed[tid], sRed[tid + s]);
        __syncthreads();
    }
    float mx = sRed[0]; __syncthreads();
    float ex = __expf(e - mx);
    sRed[tid] = ex; __syncthreads();
    for (int s = 128; s > 0; s >>= 1) {
        if (tid < s) sRed[tid] += sRed[tid + s];
        __syncthreads();
    }
    float wv = ex / sRed[0];
    sWt[l] = wv;
    wsOut[((size_t)b * 800 + t) * 256 + l] = wv;
    prevW[b * 256 + l] = wv;
    cumW[b * 256 + l]  = sCum[l] + wv;
    __syncthreads();
    for (int e0 = tid; e0 < 512; e0 += 256) {
        float a = 0.f;
        for (int l2 = 0; l2 < 256; ++l2)
            a += sWt[l2] * enc[((size_t)b * 256 + l2) * 512 + e0];
        ctx[b * 512 + e0] = a;
        int hf, jf;
        fragPos(e0 & 31, hf, jf);
        int laneF = hf * 16 + (b & 15);
        ctxFrag[((size_t)(e0 >> 5) * 2048) + (size_t)(b >> 4) * 512 + laneF * 16 + jf] = (__bf16)a;
    }
}

// ---------------------------------------------------------------------------
// Frame / stop projection: proto = [h_gen | ctx]; tiny GEMV per batch row.
// ---------------------------------------------------------------------------
__global__ void k_frame(const float* __restrict__ hG, const float* __restrict__ ctx,
                        const float* __restrict__ fW, const float* __restrict__ fb,
                        const float* __restrict__ sW, const float* __restrict__ sb,
                        float* __restrict__ frames, float* __restrict__ stops, int t)
{
    __shared__ float sP[1536];
    int b = blockIdx.x, tid = threadIdx.x;
    for (int i = tid; i < 1536; i += 128)
        sP[i] = (i < 1024) ? hG[b * 1024 + i] : ctx[b * 512 + (i - 1024)];
    __syncthreads();
    if (tid < 80) {
        float a = fb[tid];
        for (int k = 0; k < 1536; ++k) a += sP[k] * fW[k * 80 + tid];
        frames[((size_t)b * 800 + t) * 80 + tid] = a;
    } else if (tid == 80) {
        float a = sb[0];
        for (int k = 0; k < 1536; ++k) a += sP[k] * sW[k];
        stops[(size_t)b * 800 + t] = a;
    }
}

// Build prenet input: row (t*64+b), k<80 -> target[b][k][t-1]; t==0 row zero.
__global__ void k_build_xpre(const float* __restrict__ tgt, __bf16* __restrict__ xp)
{
    int idx = blockIdx.x * 256 + threadIdx.x;
    if (idx >= 51200 * 96) return;
    int k = idx % 96;
    int row = idx / 96;
    int t = row >> 6, b = row & 63;
    float v = 0.f;
    if (t > 0 && k < 80) v = tgt[((size_t)b * 80 + k) * 800 + (t - 1)];
    xp[idx] = (__bf16)v;
}

__global__ void k_zero(float* __restrict__ p, int n)
{
    int i = blockIdx.x * 256 + threadIdx.x;
    if (i < n) p[i] = 0.f;
}

// ---------------------------------------------------------------------------
extern "C" void kernel_launch(void* const* d_in, const int* in_sizes, int n_in,
                              void* d_out, int out_size, void* d_ws, size_t ws_size,
                              hipStream_t stream)
{
    const float* enc     = (const float*)d_in[0];
    const float* target  = (const float*)d_in[1];
    const int*   lengths = (const int*)  d_in[2];
    const float* pre_W1  = (const float*)d_in[3];
    const float* pre_b1  = (const float*)d_in[4];
    const float* pre_W2  = (const float*)d_in[5];
    const float* pre_b2  = (const float*)d_in[6];
    const float* att_Wih = (const float*)d_in[7];
    const float* att_Whh = (const float*)d_in[8];
    const float* att_b   = (const float*)d_in[9];
    const float* gen_Wih = (const float*)d_in[10];
    const float* gen_Whh = (const float*)d_in[11];
    const float* gen_b   = (const float*)d_in[12];
    const float* mem_W   = (const float*)d_in[13];
    const float* q_W     = (const float*)d_in[14];
    const float* loc_k   = (const float*)d_in[15];
    const float* loc_W   = (const float*)d_in[16];
    const float* vvec    = (const float*)d_in[17];
    const float* frame_W = (const float*)d_in[18];
    const float* frame_b = (const float*)d_in[19];
    const float* stop_W  = (const float*)d_in[20];
    const float* stop_b  = (const float*)d_in[21];

    float* outFrames = (float*)d_out;
    float* outStops  = outFrames + (size_t)64 * 800 * 80;
    float* outWs     = outStops  + (size_t)64 * 800;

    char* wsp = (char*)d_ws;
    auto take = [&](size_t bytes) -> char* {
        char* p = wsp;
        wsp += (bytes + 255) & ~(size_t)255;
        return p;
    };
    __bf16* attWp  = (__bf16*)take((size_t)4096 * 1792 * 2);
    __bf16* genWp  = (__bf16*)take((size_t)4096 * 2560 * 2);
    __bf16* pW1p   = (__bf16*)take((size_t)256 * 96 * 2);
    __bf16* pW2p   = (__bf16*)take((size_t)256 * 256 * 2);
    __bf16* memWp  = (__bf16*)take((size_t)128 * 512 * 2);
    __bf16* xpre   = (__bf16*)take((size_t)51200 * 96 * 2);
    __bf16* h1     = (__bf16*)take((size_t)51200 * 256 * 2);
    __bf16* xFrag  = (__bf16*)take((size_t)800 * 8 * 2048 * 2);   // prenet frags
    float*  pmem   = (float*) take((size_t)16384 * 128 * 4);

    const int stateFloats = 64 * 1024 * 4 + 64 * 512 + 64 * 256 * 2;
    float* stF   = (float*)take((size_t)stateFloats * 4);
    float* cA    = stF;
    float* cG    = cA  + 64 * 1024;
    float* hArow = cG  + 64 * 1024;
    float* hGrow = hArow + 64 * 1024;
    float* ctx   = hGrow + 64 * 1024;
    float* prevW = ctx + 64 * 512;
    float* cumW  = prevW + 64 * 256;

    const int fragBf16 = 32 * 2048 * 4 + 16 * 2048;   // hA x2, hG x2, ctx
    __bf16* fr    = (__bf16*)take((size_t)fragBf16 * 2);
    __bf16* hAf0  = fr;
    __bf16* hAf1  = hAf0 + 32 * 2048;
    __bf16* hGf0  = hAf1 + 32 * 2048;
    __bf16* hGf1  = hGf0 + 32 * 2048;
    __bf16* ctxF  = hGf1 + 32 * 2048;

    auto nb = [](size_t n) { return (unsigned)((n + 255) / 256); };

    // Pack weights (concat Wih||Whh along K for the two LSTMs).
    k_pack<<<nb((size_t)256 * 24 * 32), 256, 0, stream>>>(att_Wih, 0,  768, 4096, attWp, 56,  0, 24, 256);
    k_pack<<<nb((size_t)256 * 32 * 32), 256, 0, stream>>>(att_Whh, 0, 1024, 4096, attWp, 56, 24, 32, 256);
    k_pack<<<nb((size_t)256 * 48 * 32), 256, 0, stream>>>(gen_Wih, 0, 1536, 4096, genWp, 80,  0, 48, 256);
    k_pack<<<nb((size_t)256 * 32 * 32), 256, 0, stream>>>(gen_Whh, 0, 1024, 4096, genWp, 80, 48, 32, 256);
    k_pack<<<nb((size_t)16 * 3 * 32),  256, 0, stream>>>(pre_W1, 1,  80, 256, pW1p,  3, 0,  3, 16);
    k_pack<<<nb((size_t)16 * 8 * 32),  256, 0, stream>>>(pre_W2, 1, 256, 256, pW2p,  8, 0,  8, 16);
    k_pack<<<nb((size_t)8 * 16 * 32),  256, 0, stream>>>(mem_W,  1, 512, 128, memWp, 16, 0, 16,  8);

    // Prenet over all frames (independent of recurrence) + memory projection.
    k_build_xpre<<<nb((size_t)51200 * 96), 256, 0, stream>>>(target, xpre);
    k_gemm<__bf16><<<dim3(800, 8), 256, 0, stream>>>(xpre,  96, pW1p, 256, pre_b1, 1,
                                                     nullptr, h1, nullptr, 0);
    k_gemm<__bf16><<<dim3(800, 8), 256, 0, stream>>>(h1,   256, pW2p, 256, pre_b2, 1,
                                                     nullptr, nullptr, xFrag, 8);
    k_gemm<float><<<dim3(256, 4), 256, 0, stream>>>(enc,   512, memWp, 128, nullptr, 0,
                                                    pmem, nullptr, nullptr, 0);

    // Zero recurrent state + fragment state every call (deterministic).
    k_zero<<<nb((size_t)stateFloats), 256, 0, stream>>>(stF, stateFloats);
    k_zero<<<nb((size_t)fragBf16 / 2), 256, 0, stream>>>((float*)fr, fragBf16 / 2);

    int p = 0;
    for (int t = 0; t < 800; ++t) {
        const __bf16* xa = xFrag + (size_t)t * 8 * 2048;
        __bf16* hAin  = p ? hAf1 : hAf0;
        __bf16* hAout = p ? hAf0 : hAf1;
        __bf16* hGin  = p ? hGf1 : hGf0;
        __bf16* hGout = p ? hGf0 : hGf1;
        k_lstm<<<64, 128, 0, stream>>>(xa, 8, ctxF, 16, hAin, attWp, 56, att_b,
                                       cA, hArow, hAout);
        k_attn<<<64, 256, 0, stream>>>(hArow, enc, pmem, lengths, q_W, loc_k, loc_W,
                                       vvec, prevW, cumW, ctx, ctxF, outWs, t);
        k_lstm<<<64, 128, 0, stream>>>(hAout, 32, ctxF, 16, hGin, genWp, 80, gen_b,
                                       cG, hGrow, hGout);
        k_frame<<<64, 128, 0, stream>>>(hGrow, ctx, frame_W, frame_b, stop_W, stop_b,
                                        outFrames, outStops, t);
        p ^= 1;
    }
    (void)in_sizes; (void)n_in; (void)out_size; (void)ws_size;
}
